// SelfMultiheadAttentionAug_7146825580820
// MI455X (gfx1250) — matrix operand order repaired and checked by reading, not compiled, not run on hardware
//
#include <hip/hip_runtime.h>

// ---------------------------------------------------------------------------
// Self multi-head attention, B=2 S=2048 E=512 H=8 D=64, bf16 WMMA pipeline.
// Async LDS staging (double-buffered) of K/V chunks in flash attention.
// ---------------------------------------------------------------------------

typedef __attribute__((ext_vector_type(16))) __bf16 v16bf;
typedef __attribute__((ext_vector_type(8)))  __bf16 v8bf;
typedef __attribute__((ext_vector_type(8)))  float  v8f;
typedef __attribute__((ext_vector_type(4)))  int    v4i;

static constexpr int Bc = 2;
static constexpr int Sc = 2048;
static constexpr int Ec = 512;
static constexpr int Hc = 8;
static constexpr int Dc = 64;
static constexpr float NEG_INF = -1.0e30f;

#if defined(__has_builtin)
#if __has_builtin(__builtin_amdgcn_global_load_async_to_lds_b128)
#define HAVE_ASYNC_LDS 1
#endif
#if __has_builtin(__builtin_amdgcn_s_wait_asynccnt)
#define HAVE_WAIT_ASYNC 1
#endif
#endif

__device__ __forceinline__ __bf16 f2bf(float f) {
    unsigned u = __builtin_bit_cast(unsigned, f);
    u += 0x7FFFu + ((u >> 16) & 1u);          // round-to-nearest-even
    unsigned short h = (unsigned short)(u >> 16);
    return __builtin_bit_cast(__bf16, h);
}

__device__ __forceinline__ v8f wmma_bf16(v16bf a, v16bf b, v8f c) {
    return __builtin_amdgcn_wmma_f32_16x16x32_bf16(
        /*neg_a=*/false, a, /*neg_b=*/false, b,
        /*c_mod=*/(short)0, c, /*reuse_a=*/false, /*reuse_b=*/false);
}

// A-fragment (16x32 bf16, MxK): lane holds row M=lane%16, K elements
// {koff..koff+15} with koff = (lane&16)?16:0.
__device__ __forceinline__ v16bf load_a_frag(const __bf16* p) {
    v8bf c0 = *(const v8bf*)(p);
    v8bf c1 = *(const v8bf*)(p + 8);
    v16bf a;
#pragma unroll
    for (int i = 0; i < 8; ++i) { a[i] = c0[i]; a[8 + i] = c1[i]; }
    return a;
}

// B-fragment (32x16 bf16, KxN): lane holds column N=lane%16, contiguous
// K = 0..15 (lanes 0-15) or K = 16..31 (lanes 16-31).  p must already point
// at the column's K-major data + ((lane&16)?16:0).  32B-aligned load.
__device__ __forceinline__ v16bf load_b_frag(const __bf16* p) {
    return *(const v16bf*)(p);
}

// 16-byte global -> LDS copy: async (ASYNCcnt) when available, else direct.
__device__ __forceinline__ void copy16_g2l(const __bf16* g, __bf16* l) {
#if HAVE_ASYNC_LDS
    __builtin_amdgcn_global_load_async_to_lds_b128((v4i*)g, (v4i*)l, 0, 0);
#else
    *(v8bf*)l = *(const v8bf*)g;
#endif
}

__device__ __forceinline__ void wait_async_copies() {
#if HAVE_ASYNC_LDS
#if HAVE_WAIT_ASYNC
    __builtin_amdgcn_s_wait_asynccnt(0);
#else
    asm volatile("s_wait_asynccnt 0x0" ::: "memory");
#endif
#endif
}

// ---------------------------------------------------------------------------
// Kernel 1: fp32 -> bf16 conversions (query, in_proj_w, out_proj_w)
// ---------------------------------------------------------------------------
__global__ void cvt_kernel(const float* __restrict__ q,
                           const float* __restrict__ w1,
                           const float* __restrict__ w2,
                           __bf16* __restrict__ qb,
                           __bf16* __restrict__ w1b,
                           __bf16* __restrict__ w2b) {
    const int n0 = Bc * Sc * Ec;       // 2097152
    const int n1 = 3 * Ec * Ec;        // 786432
    const int n2 = Ec * Ec;            // 262144
    const int stride = gridDim.x * blockDim.x;
    for (int idx = blockIdx.x * blockDim.x + threadIdx.x;
         idx < n0 + n1 + n2; idx += stride) {
        if (idx < n0)            qb[idx]            = f2bf(q[idx]);
        else if (idx < n0 + n1)  w1b[idx - n0]      = f2bf(w1[idx - n0]);
        else                     w2b[idx - n0 - n1] = f2bf(w2[idx - n0 - n1]);
    }
}

// ---------------------------------------------------------------------------
// Kernel 2: QKV projection  qkv[n,m] = sum_k x[n,k]*W[m,k] + b[m]
//   q part (m<512) scaled by D^-1/2; also emits vT[b,h,d,s] for the PV GEMM.
// ---------------------------------------------------------------------------
__global__ __launch_bounds__(256) void qkv_proj_kernel(
    const __bf16* __restrict__ xb, const __bf16* __restrict__ wb,
    const float* __restrict__ bias, __bf16* __restrict__ qkv,
    __bf16* __restrict__ vT) {
    const int lane  = threadIdx.x & 31;
    const int wave  = threadIdx.x >> 5;
    const int ln    = lane & 15;
    const int koffA = (lane & 16) ? 16 : 0;
    const int koffB = (lane & 16) ? 16 : 0;
    const int row0  = blockIdx.x * 128 + wave * 16;
    const int col0  = blockIdx.y * 64;

    v8f acc[4] = {};
    const __bf16* arow = xb + (size_t)(row0 + ln) * Ec;
#pragma unroll 4
    for (int kk = 0; kk < Ec; kk += 32) {
        v16bf a = load_a_frag(arow + kk + koffA);
#pragma unroll
        for (int j = 0; j < 4; ++j) {
            const __bf16* brow = wb + (size_t)(col0 + 16 * j + ln) * Ec;
            v16bf b = load_b_frag(brow + kk + koffB);
            acc[j] = wmma_bf16(a, b, acc[j]);
        }
    }

    const int rbase = row0 + ((lane & 16) ? 8 : 0);
#pragma unroll
    for (int j = 0; j < 4; ++j) {
        const int m  = col0 + 16 * j + ln;
        const float bm = bias[m];
        const float sc = (m < Ec) ? 0.125f : 1.0f;   // D^-0.5 on q
#pragma unroll
        for (int r = 0; r < 8; ++r) {
            const int n = rbase + r;
            const float v = (acc[j][r] + bm) * sc;
            const __bf16 h = f2bf(v);
            qkv[(size_t)n * (3 * Ec) + m] = h;
            if (m >= 2 * Ec) {                        // V part -> vT[b,h,d,s]
                const int mm = m - 2 * Ec;
                const int hh = mm >> 6, d = mm & 63;
                const int bb = n >> 11, s = n & 2047;
                vT[(((size_t)(bb * Hc + hh) * Dc + d) * Sc) + s] = h;
            }
        }
    }
}

// ---------------------------------------------------------------------------
// Kernel 3: flash attention.  Block = 4 waves = 64 q rows, all sharing (b,h).
//   K/V chunks (32 keys) staged cooperatively into LDS with async copies,
//   double buffered; scores + online softmax + PV per wave with WMMA.
// ---------------------------------------------------------------------------
__global__ __launch_bounds__(128) void attn_kernel(
    const __bf16* __restrict__ qkv, const __bf16* __restrict__ vT,
    const float* __restrict__ bias, const unsigned char* __restrict__ kpm,
    __bf16* __restrict__ aout) {
    __shared__ __attribute__((aligned(64))) __bf16 kl[2][32][64];   // 8 KB
    __shared__ __attribute__((aligned(64))) __bf16 vl[2][64][32];   // 8 KB
    __shared__ __attribute__((aligned(64))) __bf16 ptile[4][16 * 32]; // 4 KB

    const int tid   = threadIdx.x;
    const int lane  = tid & 31;
    const int wave  = tid >> 5;
    const int ln    = lane & 15;
    const int koffA = (lane & 16) ? 16 : 0;
    const int koffB = (lane & 16) ? 16 : 0;
    const int rbase = (lane & 16) ? 8 : 0;

    const int gw    = blockIdx.x * 4 + wave;
    const int bh    = gw >> 7;                 // 128 strips per (b,h)
    const int strip = gw & 127;
    const int b     = bh >> 3, h = bh & 7;
    const int q0    = strip * 16;

    const __bf16* qbase = qkv + ((size_t)(b * Sc + q0 + ln)) * (3 * Ec) + h * Dc;
    const __bf16* kblk  = qkv + (size_t)(b * Sc) * (3 * Ec) + h * Dc + Ec;
    const __bf16* vtb   = vT + (size_t)(b * Hc + h) * Dc * Sc;
    const float*  bb    = bias + (size_t)bh * Sc * Sc;
    const unsigned char* mk = kpm + b * Sc;

    // q fragments (reused for every key chunk); K = head dim, two steps of 32
    const v16bf qa0 = load_a_frag(qbase + 0  + koffA);
    const v16bf qa1 = load_a_frag(qbase + 32 + koffA);

    float mrow[8], lrow[8];
#pragma unroll
    for (int r = 0; r < 8; ++r) { mrow[r] = -3.0e38f; lrow[r] = 0.0f; }
    v8f oacc[4] = {};

    // cooperative stage of one 32-key K/V chunk into LDS buffer `bf`
    auto stage_kv = [&](int bf, int k0) {
#pragma unroll
        for (int s = tid; s < 256; s += 128) {          // K: 32 rows x 8 segs
            const int kr = s >> 3, off = (s & 7) * 8;
            copy16_g2l(kblk + (size_t)(k0 + kr) * (3 * Ec) + off,
                       &kl[bf][kr][off]);
        }
#pragma unroll
        for (int s = tid; s < 256; s += 128) {          // V: 64 rows x 4 segs
            const int d = s >> 2, off = (s & 3) * 8;
            copy16_g2l(vtb + (size_t)d * Sc + k0 + off, &vl[bf][d][off]);
        }
    };

    stage_kv(0, 0);                                     // prologue

    for (int kc = 0; kc < Sc / 32; ++kc) {
        const int k0 = kc * 32;
        const int bf = kc & 1;

        wait_async_copies();      // own async issues landed in LDS
        __syncthreads();          // everyone's issues landed; prev buf free

        if (kc + 1 < Sc / 32) {
            stage_kv(bf ^ 1, k0 + 32);                  // overlap next chunk
            __builtin_prefetch(bb + (size_t)(q0 + rbase) * Sc + k0 + 32, 0, 1);
        }

        // ---- scores: 16x32 = q(16x64) x k^T(64x32), K/V read from LDS ----
        v8f s0 = {}, s1 = {};
        s0 = wmma_bf16(qa0, load_b_frag(&kl[bf][ln]     [ 0 + koffB]), s0);
        s0 = wmma_bf16(qa1, load_b_frag(&kl[bf][ln]     [32 + koffB]), s0);
        s1 = wmma_bf16(qa0, load_b_frag(&kl[bf][16 + ln][ 0 + koffB]), s1);
        s1 = wmma_bf16(qa1, load_b_frag(&kl[bf][16 + ln][32 + koffB]), s1);

        // ---- mask + bias + online softmax ----
        const int c0 = k0 + ln, c1 = k0 + 16 + ln;
        const bool msk0 = mk[c0] != 0, msk1 = mk[c1] != 0;
#pragma unroll
        for (int r = 0; r < 8; ++r) {
            const int q = q0 + rbase + r;
            float x0 = msk0 ? NEG_INF : s0[r];
            float x1 = msk1 ? NEG_INF : s1[r];
            x0 += bb[(size_t)q * Sc + c0];
            x1 += bb[(size_t)q * Sc + c1];

            float t = fmaxf(x0, x1);             // row max over 32 cols
            t = fmaxf(t, __shfl_xor(t, 1, 32));
            t = fmaxf(t, __shfl_xor(t, 2, 32));
            t = fmaxf(t, __shfl_xor(t, 4, 32));
            t = fmaxf(t, __shfl_xor(t, 8, 32));

            const float mnew = fmaxf(mrow[r], t);
            const float sc   = __expf(mrow[r] - mnew);
            const float e0   = __expf(x0 - mnew);
            const float e1   = __expf(x1 - mnew);
            float rs = e0 + e1;                  // row sum over 32 cols
            rs += __shfl_xor(rs, 1, 32);
            rs += __shfl_xor(rs, 2, 32);
            rs += __shfl_xor(rs, 4, 32);
            rs += __shfl_xor(rs, 8, 32);
            lrow[r] = lrow[r] * sc + rs;
            mrow[r] = mnew;

            ptile[wave][(rbase + r) * 32 + ln]      = f2bf(e0);
            ptile[wave][(rbase + r) * 32 + 16 + ln] = f2bf(e1);
#pragma unroll
            for (int jt = 0; jt < 4; ++jt) oacc[jt][r] *= sc;
        }

        asm volatile("s_wait_dscnt 0x0" ::: "memory");  // P tile visible

        // ---- PV: o += P(16x32) x V(32x64) ----
        const v16bf pa = load_a_frag(&ptile[wave][ln * 32 + koffA]);
#pragma unroll
        for (int jt = 0; jt < 4; ++jt)
            oacc[jt] = wmma_bf16(pa, load_b_frag(&vl[bf][16 * jt + ln][koffB]),
                                 oacc[jt]);

        __syncthreads();          // all waves done with buf before refill
    }

    // ---- normalize + store (bf16, [b,s,e] layout) ----
#pragma unroll
    for (int jt = 0; jt < 4; ++jt) {
#pragma unroll
        for (int r = 0; r < 8; ++r) {
            const int q = q0 + rbase + r;
            const float v = oacc[jt][r] / lrow[r];
            aout[((size_t)(b * Sc + q)) * Ec + h * Dc + 16 * jt + ln] = f2bf(v);
        }
    }
}

// ---------------------------------------------------------------------------
// Kernel 4: output projection  out[n,m] = sum_k a[n,k]*Wo[m,k] + bo[m]  (fp32)
// ---------------------------------------------------------------------------
__global__ __launch_bounds__(256) void out_proj_kernel(
    const __bf16* __restrict__ ab, const __bf16* __restrict__ wb,
    const float* __restrict__ bias, float* __restrict__ out) {
    const int lane  = threadIdx.x & 31;
    const int wave  = threadIdx.x >> 5;
    const int ln    = lane & 15;
    const int koffA = (lane & 16) ? 16 : 0;
    const int koffB = (lane & 16) ? 16 : 0;
    const int row0  = blockIdx.x * 128 + wave * 16;
    const int col0  = blockIdx.y * 64;

    v8f acc[4] = {};
    const __bf16* arow = ab + (size_t)(row0 + ln) * Ec;
#pragma unroll 4
    for (int kk = 0; kk < Ec; kk += 32) {
        v16bf a = load_a_frag(arow + kk + koffA);
#pragma unroll
        for (int j = 0; j < 4; ++j) {
            const __bf16* brow = wb + (size_t)(col0 + 16 * j + ln) * Ec;
            acc[j] = wmma_bf16(a, load_b_frag(brow + kk + koffB), acc[j]);
        }
    }

    const int rbase = row0 + ((lane & 16) ? 8 : 0);
#pragma unroll
    for (int j = 0; j < 4; ++j) {
        const int m = col0 + 16 * j + ln;
        const float bm = bias[m];
#pragma unroll
        for (int r = 0; r < 8; ++r)
            out[(size_t)(rbase + r) * Ec + m] = acc[j][r] + bm;
    }
}

// ---------------------------------------------------------------------------
// Host launcher
// ---------------------------------------------------------------------------
extern "C" void kernel_launch(void* const* d_in, const int* in_sizes, int n_in,
                              void* d_out, int out_size, void* d_ws, size_t ws_size,
                              hipStream_t stream) {
    (void)in_sizes; (void)n_in; (void)out_size; (void)ws_size;
    const float* query  = (const float*)d_in[0];
    const unsigned char* kpm = (const unsigned char*)d_in[1];  // bool mask
    const float* abias  = (const float*)d_in[2];
    const float* b_in   = (const float*)d_in[4];
    const float* w_in   = (const float*)d_in[3];
    const float* w_out  = (const float*)d_in[5];
    const float* b_out  = (const float*)d_in[6];

    __bf16* qb   = (__bf16*)d_ws;                      // 4096*512
    __bf16* w1b  = qb   + (size_t)Bc * Sc * Ec;        // 1536*512
    __bf16* w2b  = w1b  + (size_t)3 * Ec * Ec;         // 512*512
    __bf16* qkv  = w2b  + (size_t)Ec * Ec;             // 4096*1536
    __bf16* vT   = qkv  + (size_t)Bc * Sc * 3 * Ec;    // 16*64*2048
    __bf16* aout = vT   + (size_t)Bc * Hc * Dc * Sc;   // 4096*512

    cvt_kernel<<<3072, 256, 0, stream>>>(query, w_in, w_out, qb, w1b, w2b);

    qkv_proj_kernel<<<dim3((Bc * Sc) / 128, (3 * Ec) / 64), 256, 0, stream>>>(
        qb, w1b, b_in, qkv, vT);

    attn_kernel<<<(Bc * Hc * (Sc / 16)) / 4, 128, 0, stream>>>(
        qkv, vT, abias, kpm, aout);

    out_proj_kernel<<<dim3((Bc * Sc) / 128, Ec / 64), 256, 0, stream>>>(
        aout, w2b, b_out, (float*)d_out);
}
